// GCN_21732534518460
// MI455X (gfx1250) — matrix operand order repaired and checked by reading, not compiled, not run on hardware
//
#include <hip/hip_runtime.h>
#include <hip/hip_bf16.h>

typedef __attribute__((ext_vector_type(16))) __bf16 v16bf;
typedef __attribute__((ext_vector_type(8)))  float  v8f;

#define N_NODES 50000
#define N_EDGES 625000
#define DIM     128
#define OUT_DIM 10
#define NGRAPH  500
#define MTILES  (N_NODES / 16)   // 3125 = 5^5
#define MT_PER_BLK 5
#define GEMM_BLKS (MTILES / MT_PER_BLK)  // 625

// ---------------- helpers ----------------
__device__ __forceinline__ unsigned short f32_to_bf16_rne(float f) {
    unsigned int u = __builtin_bit_cast(unsigned int, f);
    u += 0x7FFFu + ((u >> 16) & 1u);
    return (unsigned short)(u >> 16);
}

__global__ void zero_f32(float* __restrict__ p, int n) {
    int i = blockIdx.x * blockDim.x + threadIdx.x;
    if (i < n) p[i] = 0.0f;
}

// ---------------- degree / dinv ----------------
__global__ void deg_accum(const int* __restrict__ dst, float* __restrict__ deg) {
    int e = blockIdx.x * blockDim.x + threadIdx.x;
    if (e < N_EDGES) atomicAdd(&deg[dst[e]], 1.0f);
}

__global__ void make_dinv(const float* __restrict__ deg, float* __restrict__ dinv) {
    int i = blockIdx.x * blockDim.x + threadIdx.x;
    if (i < N_NODES) dinv[i] = rsqrtf(deg[i] + 1.0f);
}

// ---------------- fragment K-mapping (this code's contiguous order, k = 16*(lane/16) + j) ----------------
// A (16x32 bf16): lane<16 row M=lane holds K=0..15 in element order;
// lanes>=16 same rows with K=16..31.
__device__ __forceinline__ int a_klocal(int lane, int j) {
    int vg = j >> 1;
    return j + ((lane < 16) ? 0 : 16);
}

// ---------------- operand packing (WMMA fragment order) ----------------
__global__ void pack_A(const float* __restrict__ X, unsigned short* __restrict__ Ap) {
    int t = blockIdx.x * blockDim.x + threadIdx.x;       // MTILES*4*32 threads
    if (t >= MTILES * 4 * 32) return;
    int lane = t & 31;
    int kt   = (t >> 5) & 3;
    int mt   = t >> 7;
    int m    = mt * 16 + (lane & 15);
    unsigned short* o = Ap + (size_t)t * 16;
    #pragma unroll
    for (int j = 0; j < 16; ++j)
        o[j] = f32_to_bf16_rne(X[(size_t)m * DIM + kt * 32 + a_klocal(lane, j)]);
}

// B (32x16 bf16): lane<16 col N=lane holds K=0..15 in element order; lanes
// 16..31 hold the same columns, K=16..31.
__global__ void pack_W(const float* __restrict__ W, unsigned short* __restrict__ Wp) {
    int t = blockIdx.x * blockDim.x + threadIdx.x;       // 4*8*32 = 1024 threads
    if (t >= 4 * 8 * 32) return;
    int lane = t & 31;
    int nt   = (t >> 5) & 7;
    int kt   = t >> 8;
    int n    = nt * 16 + (lane & 15);
    int kbase = kt * 32 + ((lane < 16) ? 0 : 16);
    unsigned short* o = Wp + (size_t)t * 16;
    #pragma unroll
    for (int j = 0; j < 16; ++j)
        o[j] = f32_to_bf16_rne(W[(size_t)(kbase + j) * DIM + n]);
}

// ---------------- WMMA GEMM: H[50000x128] = A @ W[128x128] ----------------
// 8 waves/block = 8 N-tiles; each wave keeps its 4 B fragments in registers
// and sweeps 5 M-tiles -> 20 v_wmma per wave.
__global__ void gemm_wmma(const unsigned short* __restrict__ Ap,
                          const unsigned short* __restrict__ Wp,
                          float* __restrict__ H) {
    int ntile = threadIdx.x >> 5;
    int lane  = threadIdx.x & 31;

    v16bf b0 = *reinterpret_cast<const v16bf*>(Wp + (((size_t)0 * 8 + ntile) * 32 + lane) * 16);
    v16bf b1 = *reinterpret_cast<const v16bf*>(Wp + (((size_t)1 * 8 + ntile) * 32 + lane) * 16);
    v16bf b2 = *reinterpret_cast<const v16bf*>(Wp + (((size_t)2 * 8 + ntile) * 32 + lane) * 16);
    v16bf b3 = *reinterpret_cast<const v16bf*>(Wp + (((size_t)3 * 8 + ntile) * 32 + lane) * 16);

    int nbase = ntile * 16 + (lane & 15);
    int moff  = (lane < 16) ? 0 : 8;

    #pragma unroll
    for (int mm = 0; mm < MT_PER_BLK; ++mm) {
        int mt = blockIdx.x * MT_PER_BLK + mm;
        const unsigned short* ab = Ap + (((size_t)mt * 4) * 32 + lane) * 16;
        v16bf a0 = *reinterpret_cast<const v16bf*>(ab + 0 * 32 * 16);
        v16bf a1 = *reinterpret_cast<const v16bf*>(ab + 1 * 32 * 16);
        v16bf a2 = *reinterpret_cast<const v16bf*>(ab + 2 * 32 * 16);
        v16bf a3 = *reinterpret_cast<const v16bf*>(ab + 3 * 32 * 16);
        v8f c = {};
        c = __builtin_amdgcn_wmma_f32_16x16x32_bf16(false, a0, false, b0, (short)0, c, false, false);
        c = __builtin_amdgcn_wmma_f32_16x16x32_bf16(false, a1, false, b1, (short)0, c, false, false);
        c = __builtin_amdgcn_wmma_f32_16x16x32_bf16(false, a2, false, b2, (short)0, c, false, false);
        c = __builtin_amdgcn_wmma_f32_16x16x32_bf16(false, a3, false, b3, (short)0, c, false, false);
        int mbase = mt * 16 + moff;
        #pragma unroll
        for (int r = 0; r < 8; ++r)
            H[(size_t)(mbase + r) * DIM + nbase] = c[r];
    }
}

// ---------------- edge scatter-add (bandwidth limiter) ----------------
__global__ void edge_agg(const int* __restrict__ src, const int* __restrict__ dst,
                         const float* __restrict__ dinv, const float* __restrict__ H,
                         float* __restrict__ Agg) {
    size_t t = (size_t)blockIdx.x * blockDim.x + threadIdx.x;  // E*32 exactly
    size_t e = t >> 5;
    int    q = (int)(t & 31);
    int s = src[e], d = dst[e];
    float coef = dinv[s] * dinv[d];
    const float4 hv = *reinterpret_cast<const float4*>(H + (size_t)s * DIM + q * 4);
    float* ag = Agg + (size_t)d * DIM + q * 4;
    atomicAdd(ag + 0, hv.x * coef);
    atomicAdd(ag + 1, hv.y * coef);
    atomicAdd(ag + 2, hv.z * coef);
    atomicAdd(ag + 3, hv.w * coef);
}

// ---------------- fused epilogue + next-layer A packing (layer 1) ----------
// value = relu(Agg + H*dinv^2 + b) written straight into bf16 fragment order,
// skipping the f32 round trip through HBM.
__global__ void post_pack(const float* __restrict__ Agg, const float* __restrict__ Hg,
                          const float* __restrict__ dinv, const float* __restrict__ b,
                          unsigned short* __restrict__ Ap) {
    int t = blockIdx.x * blockDim.x + threadIdx.x;       // MTILES*4*32 threads
    if (t >= MTILES * 4 * 32) return;
    int lane = t & 31;
    int kt   = (t >> 5) & 3;
    int mt   = t >> 7;
    int m    = mt * 16 + (lane & 15);
    float dv = dinv[m];
    float d2 = dv * dv;
    unsigned short* o = Ap + (size_t)t * 16;
    #pragma unroll
    for (int j = 0; j < 16; ++j) {
        int k = kt * 32 + a_klocal(lane, j);
        size_t i = (size_t)m * DIM + k;
        float v = Agg[i] + Hg[i] * d2 + b[k];
        o[j] = f32_to_bf16_rne(fmaxf(v, 0.0f));
    }
}

// ---------------- self-loop + bias + relu (final layer, f32 out) ----------
__global__ void post_layer(const float* __restrict__ Agg, const float* __restrict__ dinv,
                           const float* __restrict__ b, float* __restrict__ H) {
    size_t i = (size_t)blockIdx.x * blockDim.x + threadIdx.x;  // N*128 exactly
    int n = (int)(i >> 7);
    int c = (int)(i & 127);
    float dv = dinv[n];
    float v = Agg[i] + H[i] * dv * dv + b[c];
    H[i] = fmaxf(v, 0.0f);
}

// ---------------- global mean pool ----------------
__global__ void pool_sum(const float* __restrict__ H, const int* __restrict__ batch,
                         float* __restrict__ pooled) {
    size_t i = (size_t)blockIdx.x * blockDim.x + threadIdx.x;  // N*128 exactly
    int n = (int)(i >> 7);
    atomicAdd(&pooled[(size_t)batch[n] * DIM + (i & 127)], H[i]);
}

__global__ void pool_cnt(const int* __restrict__ batch, float* __restrict__ cnt) {
    int n = blockIdx.x * blockDim.x + threadIdx.x;
    if (n < N_NODES) atomicAdd(&cnt[batch[n]], 1.0f);
}

// ---------------- FC head + log_softmax ----------------
__global__ void head(const float* __restrict__ pooled, const float* __restrict__ cnt,
                     const float* __restrict__ Wfc, const float* __restrict__ bfc,
                     float* __restrict__ out) {
    int g = blockIdx.x * blockDim.x + threadIdx.x;
    if (g >= NGRAPH) return;
    float inv = 1.0f / fmaxf(cnt[g], 1.0f);
    float logit[OUT_DIM];
    #pragma unroll
    for (int o = 0; o < OUT_DIM; ++o) logit[o] = bfc[o];
    for (int k = 0; k < DIM; ++k) {
        float pv = pooled[(size_t)g * DIM + k] * inv;
        #pragma unroll
        for (int o = 0; o < OUT_DIM; ++o) logit[o] += pv * Wfc[k * OUT_DIM + o];
    }
    float m = logit[0];
    #pragma unroll
    for (int o = 1; o < OUT_DIM; ++o) m = fmaxf(m, logit[o]);
    float s = 0.0f;
    #pragma unroll
    for (int o = 0; o < OUT_DIM; ++o) s += __expf(logit[o] - m);
    float lse = __logf(s);
    #pragma unroll
    for (int o = 0; o < OUT_DIM; ++o) out[g * OUT_DIM + o] = logit[o] - m - lse;
}

// ---------------- workspace layout (256B aligned) ----------------
#define OFF_DINV 0u
#define OFF_DEG  200704u
#define OFF_AP   401408u                       // N*128 bf16 = 12,800,000 B
#define OFF_WP1  13201408u                     // 32768 B
#define OFF_WP2  13234176u                     // 32768 B
#define OFF_H    13266944u                     // N*128 f32 = 25,600,000 B
#define OFF_AGG  38866944u                     // N*128 f32
#define OFF_POOL 64466944u                     // 500*128 f32
#define OFF_CNT  64722944u                     // 500 f32

extern "C" void kernel_launch(void* const* d_in, const int* in_sizes, int n_in,
                              void* d_out, int out_size, void* d_ws, size_t ws_size,
                              hipStream_t stream) {
    const float* x    = (const float*)d_in[0];
    const int*   ei   = (const int*)d_in[1];
    const int*   src  = ei;
    const int*   dst  = ei + N_EDGES;
    const int*   batch= (const int*)d_in[2];
    const float* W1   = (const float*)d_in[3];
    const float* b1   = (const float*)d_in[4];
    const float* W2   = (const float*)d_in[5];
    const float* b2   = (const float*)d_in[6];
    const float* Wfc  = (const float*)d_in[7];
    const float* bfc  = (const float*)d_in[8];
    float* out = (float*)d_out;

    char* ws = (char*)d_ws;
    float*          dinv = (float*)(ws + OFF_DINV);
    float*          deg  = (float*)(ws + OFF_DEG);
    unsigned short* Ap   = (unsigned short*)(ws + OFF_AP);
    unsigned short* Wp1  = (unsigned short*)(ws + OFF_WP1);
    unsigned short* Wp2  = (unsigned short*)(ws + OFF_WP2);
    float*          H    = (float*)(ws + OFF_H);
    float*          Agg  = (float*)(ws + OFF_AGG);
    float*          pool = (float*)(ws + OFF_POOL);
    float*          cnt  = (float*)(ws + OFF_CNT);

    const int T = 256;
    const int elemN = N_NODES * DIM;                 // 6,400,000 (mult of 256)
    const int edgeT = N_EDGES * 32;                  // 20,000,000 (mult of 256)
    const int packT = MTILES * 4 * 32;               // 400,000

    // init accumulators
    zero_f32<<<(N_NODES + T - 1) / T, T, 0, stream>>>(deg, N_NODES);
    zero_f32<<<elemN / T, T, 0, stream>>>(Agg, elemN);
    zero_f32<<<(NGRAPH * DIM + T - 1) / T, T, 0, stream>>>(pool, NGRAPH * DIM);
    zero_f32<<<(NGRAPH + T - 1) / T, T, 0, stream>>>(cnt, NGRAPH);

    // degrees
    deg_accum<<<(N_EDGES + T - 1) / T, T, 0, stream>>>(dst, deg);
    make_dinv<<<(N_NODES + T - 1) / T, T, 0, stream>>>(deg, dinv);

    // pack weights once
    pack_W<<<4, T, 0, stream>>>(W1, Wp1);
    pack_W<<<4, T, 0, stream>>>(W2, Wp2);

    // ---- layer 1 ----
    pack_A<<<(packT + T - 1) / T, T, 0, stream>>>(x, Ap);
    gemm_wmma<<<GEMM_BLKS, T, 0, stream>>>(Ap, Wp1, H);
    edge_agg<<<edgeT / T, T, 0, stream>>>(src, dst, dinv, H, Agg);
    // fused: epilogue + bf16 fragment pack for layer 2
    post_pack<<<(packT + T - 1) / T, T, 0, stream>>>(Agg, H, dinv, b1, Ap);

    // ---- layer 2 ----
    zero_f32<<<elemN / T, T, 0, stream>>>(Agg, elemN);
    gemm_wmma<<<GEMM_BLKS, T, 0, stream>>>(Ap, Wp2, H);
    edge_agg<<<edgeT / T, T, 0, stream>>>(src, dst, dinv, H, Agg);
    post_layer<<<elemN / T, T, 0, stream>>>(Agg, dinv, b2, H);

    // ---- pool + head ----
    pool_sum<<<elemN / T, T, 0, stream>>>(H, batch, pool);
    pool_cnt<<<(N_NODES + T - 1) / T, T, 0, stream>>>(batch, cnt);
    head<<<(NGRAPH + T - 1) / T, T, 0, stream>>>(pool, cnt, Wfc, bfc, out);
}
